// YoloModel_73890617360567
// MI455X (gfx1250) — 8-bit matrix operand order repaired and checked by reading, compiled, not run on hardware
//
#include <hip/hip_runtime.h>

typedef __attribute__((ext_vector_type(8))) int v8i;

#define NDET    10647
#define NP      10656
#define TILES   666
#define WRDS    333
#define SORTN   16384
#define KEPTN   4096
#define OUTN    3000
#define MAXD    3000
#define CONF_T  0.8f
#define NMS_T   0.4f
#define NEGF    (-1000000000.0f)

__global__ void init_ws(unsigned long long* keys, int* counter) {
    int i = blockIdx.x * blockDim.x + threadIdx.x;
    if (i < SORTN) keys[i] = 0ull;
    if (i == 0) *counter = 0;
}

__global__ void prep_kernel(const float* __restrict__ det,
                            float4* __restrict__ boxes,
                            float* __restrict__ clsConf,
                            int* __restrict__ clsId,
                            unsigned long long* __restrict__ keys,
                            int* __restrict__ counter) {
    int wave = blockIdx.x * (blockDim.x >> 5) + (threadIdx.x >> 5);
    int lane = threadIdx.x & 31;
    if (wave >= NDET) return;
    const float* row = det + (size_t)wave * 85;
    __builtin_prefetch(row + 85, 0, 0);
    float v0 = row[lane];
    float v1 = row[lane + 32];
    float v2 = (lane < 21) ? row[lane + 64] : NEGF;

    float mv = (lane >= 5) ? v0 : NEGF;
    int   mi = lane - 5;
    if (v1 > mv) { mv = v1; mi = lane + 27; }
    if (v2 > mv) { mv = v2; mi = lane + 59; }
    for (int off = 16; off > 0; off >>= 1) {
        float ov = __shfl_xor(mv, off, 32);
        int   oi = __shfl_xor(mi, off, 32);
        if (ov > mv || (ov == mv && oi < mi)) { mv = ov; mi = oi; }
    }
    float cx  = __shfl(v0, 0, 32), cy = __shfl(v0, 1, 32);
    float w   = __shfl(v0, 2, 32), h  = __shfl(v0, 3, 32);
    float obj = __shfl(v0, 4, 32);
    if (lane == 0) {
        boxes[wave]   = make_float4(cx - w * 0.5f, cy - h * 0.5f,
                                    cx + w * 0.5f, cy + h * 0.5f);
        clsConf[wave] = mv;
        clsId[wave]   = mi;
        if (obj >= CONF_T) {
            atomicAdd(counter, 1);
            unsigned sb = __float_as_uint(obj * mv);
            keys[wave] = ((unsigned long long)sb << 32) | (unsigned)wave;
        }
    }
}

__global__ void bitonic_step(unsigned long long* keys, int k, int j) {
    int i = blockIdx.x * blockDim.x + threadIdx.x;
    int ixj = i ^ j;
    if (ixj > i) {
        unsigned long long a = keys[i], b = keys[ixj];
        if (((i & k) == 0) ? (a < b) : (a > b)) { keys[i] = b; keys[ixj] = a; }
    }
}

__global__ void gather_sorted(const unsigned long long* __restrict__ keys,
                              const float4* __restrict__ boxes,
                              const float* __restrict__ clsConf,
                              const int* __restrict__ clsId,
                              float4* __restrict__ sBox,
                              float* __restrict__ sConf,
                              int* __restrict__ sCls) {
    int r = blockIdx.x * blockDim.x + threadIdx.x;
    if (r >= NP) return;
    unsigned long long key = keys[r];
    int  idx = (int)(key & 0xFFFFFFFFull);
    bool val = (key >> 32) != 0;
    float4 pad = make_float4(-1.0e6f, -1.0e6f, -1.0e6f, -1.0e6f);
    sBox[r]  = val ? boxes[idx]   : pad;
    sConf[r] = val ? clsConf[idx] : NEGF;
    sCls[r]  = val ? clsId[idx]   : -1;
}

__device__ inline int onehot4(int cls, int K0) {
    int r = 0;
    r |= (cls == K0    ) ? 0x00000001 : 0;
    r |= (cls == K0 + 1) ? 0x00000100 : 0;
    r |= (cls == K0 + 2) ? 0x00010000 : 0;
    r |= (cls == K0 + 3) ? 0x01000000 : 0;
    return r;
}

__global__ void build_mask(const float4* __restrict__ sBox,
                           const int* __restrict__ sCls,
                           const int* __restrict__ counter,
                           unsigned short* __restrict__ mask16) {
    int bid = blockIdx.x;
    int ti = bid / TILES, tj = bid % TILES;
    int V = *counter;
    if (ti * 16 >= V || tj * 16 >= V) return;

    int lane   = threadIdx.x;
    int laneLo = lane & 15, laneHi = lane >> 4;
    int rowCls = sCls[ti * 16 + laneLo];
    int colCls = sCls[tj * 16 + laneLo];

    v8i A, B, A2, B2;
#pragma unroll
    for (int v = 0; v < 8; ++v) {
        int kA = ((v >> 2) << 5) + laneHi * 16 + ((v & 3) << 2);
        int kB = ((v >> 2) << 5) + laneHi * 16 + ((v & 3) << 2);
        A[v]  = onehot4(rowCls, kA);       B[v]  = onehot4(colCls, kB);
        A2[v] = onehot4(rowCls, kA + 64);  B2[v] = onehot4(colCls, kB + 64);
    }
    v8i acc = {0, 0, 0, 0, 0, 0, 0, 0};
    acc = __builtin_amdgcn_wmma_i32_16x16x64_iu8(false, A,  false, B,  acc, false, false);
    acc = __builtin_amdgcn_wmma_i32_16x16x64_iu8(false, A2, false, B2, acc, false, false);

    float4 cb = sBox[tj * 16 + laneLo];
    float ca = (cb.z - cb.x + 1.0f) * (cb.w - cb.y + 1.0f);
#pragma unroll
    for (int r = 0; r < 8; ++r) {
        int rowIdx = ti * 16 + r + 8 * laneHi;
        float4 rb = sBox[rowIdx];
        float ix1 = fmaxf(rb.x, cb.x), iy1 = fmaxf(rb.y, cb.y);
        float ix2 = fminf(rb.z, cb.z), iy2 = fminf(rb.w, cb.w);
        float inter = fmaxf(ix2 - ix1 + 1.0f, 0.0f) * fmaxf(iy2 - iy1 + 1.0f, 0.0f);
        float ra  = (rb.z - rb.x + 1.0f) * (rb.w - rb.y + 1.0f);
        float uni = ra + ca - inter + 1e-16f;
        bool sup = (inter > NMS_T * uni) && (acc[r] != 0);
        unsigned bal = __builtin_amdgcn_ballot_w32(sup);
        if (lane == 0) {
            mask16[(size_t)(ti * 16 + r)     * TILES + tj] = (unsigned short)(bal & 0xFFFFu);
            mask16[(size_t)(ti * 16 + r + 8) * TILES + tj] = (unsigned short)(bal >> 16);
        }
    }
}

__global__ void nms_scan(const unsigned long long* __restrict__ keys,
                         const float* __restrict__ sConf,
                         const unsigned* __restrict__ mask32,
                         float* __restrict__ out) {
    __shared__ unsigned alive[WRDS];
    __shared__ float kept[KEPTN];
    __shared__ int sFirst, sSeed;
    int tid = threadIdx.x;

    if (tid < WRDS) {
        unsigned wbits = 0;
        int base = tid * 32;
        for (int b = 0; b < 32; ++b)
            if ((keys[base + b] >> 32) != 0) wbits |= (1u << b);
        alive[tid] = wbits;
    }
    for (int i = tid; i < KEPTN; i += blockDim.x) kept[i] = NEGF;
    __syncthreads();

    int cnt = 0;
    for (int step = 0; step < MAXD; ++step) {
        if (tid == 0) sFirst = 0x7fffffff;
        __syncthreads();
        if (tid < WRDS && alive[tid] != 0) atomicMin(&sFirst, tid);
        __syncthreads();
        int fw = sFirst;
        if (fw == 0x7fffffff) break;
        if (tid == 0) {
            unsigned wv = alive[fw];
            int seed = fw * 32 + (__ffs(wv) - 1);
            sSeed = seed;
            kept[cnt++] = sConf[seed];
        }
        __syncthreads();
        int seed = sSeed;
        if (tid < WRDS && tid >= fw)
            alive[tid] &= ~mask32[(size_t)seed * WRDS + tid];
        __syncthreads();
    }
    __syncthreads();

    for (int k = 2; k <= KEPTN; k <<= 1)
        for (int j = k >> 1; j > 0; j >>= 1) {
            for (int i = tid; i < KEPTN; i += blockDim.x) {
                int ixj = i ^ j;
                if (ixj > i) {
                    float a = kept[i], b = kept[ixj];
                    if (((i & k) == 0) ? (a < b) : (a > b)) { kept[i] = b; kept[ixj] = a; }
                }
            }
            __syncthreads();
        }
    for (int i = tid; i < OUTN; i += blockDim.x) out[i] = kept[i];
}

extern "C" void kernel_launch(void* const* d_in, const int* in_sizes, int n_in,
                              void* d_out, int out_size, void* d_ws, size_t ws_size,
                              hipStream_t stream) {
    const float* det = (const float*)d_in[0];
    float* out = (float*)d_out;
    (void)in_sizes; (void)n_in; (void)out_size; (void)ws_size;

    char* ws = (char*)d_ws;
    size_t off = 0;
    auto carve = [&](size_t bytes) -> void* {
        void* p = ws + off;
        off = (off + bytes + 255) & ~(size_t)255;
        return p;
    };
    int*                counter = (int*)carve(sizeof(int));
    unsigned long long* keys    = (unsigned long long*)carve((size_t)SORTN * 8);
    float4*             boxes   = (float4*)carve((size_t)NP * 16);
    float*              clsConf = (float*)carve((size_t)NP * 4);
    int*                clsId   = (int*)carve((size_t)NP * 4);
    float4*             sBox    = (float4*)carve((size_t)NP * 16);
    float*              sConf   = (float*)carve((size_t)NP * 4);
    int*                sCls    = (int*)carve((size_t)NP * 4);
    unsigned short*     mask16  = (unsigned short*)carve((size_t)NP * TILES * 2);

    init_ws<<<SORTN / 256, 256, 0, stream>>>(keys, counter);
    prep_kernel<<<(NDET + 7) / 8, 256, 0, stream>>>(det, boxes, clsConf, clsId, keys, counter);
    for (int k = 2; k <= SORTN; k <<= 1)
        for (int j = k >> 1; j > 0; j >>= 1)
            bitonic_step<<<SORTN / 256, 256, 0, stream>>>(keys, k, j);
    gather_sorted<<<(NP + 255) / 256, 256, 0, stream>>>(keys, boxes, clsConf, clsId,
                                                        sBox, sConf, sCls);
    build_mask<<<TILES * TILES, 32, 0, stream>>>(sBox, sCls, counter, mask16);
    nms_scan<<<1, 1024, 0, stream>>>(keys, sConf, (const unsigned*)mask16, out);
}
